// EEG_GAT_35837207118112
// MI455X (gfx1250) — matrix operand order repaired and checked by reading, not compiled, not run on hardware
//
#include <hip/hip_runtime.h>
#include <hip/hip_bf16.h>

// GAT layer for MI455X (gfx1250): wave32, bf16 hi/lo split-compensated WMMA,
// branch-free b128 loads, B operands pre-packed into WMMA fragment layout.

typedef __attribute__((ext_vector_type(16))) __bf16       v16bf;
typedef __attribute__((ext_vector_type(8)))  float        v8f;
typedef __attribute__((ext_vector_type(4)))  float        v4f;
typedef __attribute__((ext_vector_type(4)))  unsigned int v4u;

#define F_DIM   200
#define NNODES  1024
#define NROWS_T 65536          // B*N
#define NTILES  13             // ceil(200/16)
#define NKT_W   7              // ceil(200/32)  k-tiles for W
#define NKT_H   32             // 1024/32       k-tiles for h0

union Frag { v16bf b; v4u u[2]; };

#define WMMA_BF16(A, B, C) \
    __builtin_amdgcn_wmma_f32_16x16x32_bf16(false, (A), false, (B), (short)0, (C), false, false)

// float[16] -> bf16 hi + bf16 residual lo
__device__ __forceinline__ void cvt_hilo(const float* v, v16bf& hi, v16bf& lo) {
#pragma unroll
    for (int i = 0; i < 16; ++i) {
        __bf16 h = (__bf16)v[i];
        hi[i] = h;
        lo[i] = (__bf16)(v[i] - (float)h);
    }
}

// A fragment (16x32 bf16, ISA 7.12.2): elements 0..7 <- arow[k0+half*8 .. +7],
// elements 8..15 <- arow[k0+16+half*8 .. +7].  Unconditional 4x b128 loads.
__device__ __forceinline__ void load_a_frag(const float* __restrict__ arow, int k0, int half,
                                            v16bf& ahi, v16bf& alo) {
    const v4f* p0 = (const v4f*)(arow + k0 + half * 8);
    const v4f* p1 = (const v4f*)(arow + k0 + 16 + half * 8);
    v4f f0 = p0[0], f1 = p0[1], f2 = p1[0], f3 = p1[1];
    float va[16];
#pragma unroll
    for (int i = 0; i < 4; ++i) { va[i] = f0[i]; va[4+i] = f1[i]; va[8+i] = f2[i]; va[12+i] = f3[i]; }
    cvt_hilo(va, ahi, alo);
}

// K-tail variant: clamped addresses (always-valid loads) + select-to-zero. No exec branches.
__device__ __forceinline__ void load_a_frag_tail(const float* __restrict__ arow, int k0, int half,
                                                 int K, v16bf& ahi, v16bf& alo) {
    float va[16];
#pragma unroll
    for (int i = 0; i < 16; ++i) {
        int k = (i < 8) ? (k0 + half * 8 + i) : (k0 + 16 + half * 8 + (i - 8));
        int kc = (k < K) ? k : (K - 1);
        float v = arow[kc];
        va[i] = (k < K) ? v : 0.0f;
    }
    cvt_hilo(va, ahi, alo);
}

// Pre-pack a B matrix (K x ncols, row-major f32) into bf16 hi/lo WMMA fragments.
// Fragment (kt,t): per lane 16 dwords = [hi: 8 dwords][lo: 8 dwords].
// B 32x16 layout: lanes 0-15 hold K = 0..7,16..23, lanes 16-31 hold K = 8..15,24..31, col = t*16 + (lane&15).
__global__ void __launch_bounds__(32) pack_b_frags(
    const float* __restrict__ B, int ldb, int K, int ncols,
    unsigned int* __restrict__ bp)
{
    int t    = blockIdx.x % NTILES;
    int kt   = blockIdx.x / NTILES;
    int lane = threadIdx.x;
    int half = lane >> 4, mlane = lane & 15;
    int n    = t * 16 + mlane;
    bool nok = n < ncols;
    int nc   = nok ? n : (ncols - 1);
    float vb[16];
#pragma unroll
    for (int i = 0; i < 16; ++i) {
        int k  = kt * 32 + ((i < 8) ? (half * 8 + i) : (16 + half * 8 + (i - 8)));
        int kc = (k < K) ? k : (K - 1);
        float v = B[kc * ldb + nc];
        vb[i] = (nok && k < K) ? v : 0.0f;        // zero-pad edges at pack time
    }
    Frag hi, lo;
    cvt_hilo(vb, hi.b, lo.b);
    v4u* dst = (v4u*)bp + ((unsigned)blockIdx.x * 32 + lane) * 4;
    dst[0] = hi.u[0]; dst[1] = hi.u[1]; dst[2] = lo.u[0]; dst[3] = lo.u[1];
}

// One wave: 16-row strip of C = A @ B using pre-packed B fragments.
// 3 bf16 WMMAs per tile per k-step: hi*hi + hi*lo + lo*hi  (~f32 accuracy).
__device__ __forceinline__ void wave_gemm_packed(
    const float* __restrict__ arow, int K, int nkt,
    const unsigned int* __restrict__ bpack, int lane, v8f acc[NTILES])
{
    const int half = lane >> 4;
#pragma unroll
    for (int t = 0; t < NTILES; ++t) acc[t] = (v8f){0.f,0.f,0.f,0.f,0.f,0.f,0.f,0.f};

    for (int kt = 0; kt < nkt; ++kt) {
        int k0 = kt * 32;
        v16bf ahi, alo;
        if (k0 + 32 <= K) load_a_frag(arow, k0, half, ahi, alo);
        else              load_a_frag_tail(arow, k0, half, K, ahi, alo);

        const v4u* fbase = (const v4u*)bpack + ((unsigned)(kt * NTILES) * 32 + lane) * 4;
#pragma unroll
        for (int t = 0; t < NTILES; ++t) {
            const v4u* f = fbase + t * 128;    // 32 lanes * 4 v4u per fragment
            Frag bh, bl;
            bh.u[0] = f[0]; bh.u[1] = f[1];    // 4x global_load_b128
            bl.u[0] = f[2]; bl.u[1] = f[3];
            acc[t] = WMMA_BF16(ahi, bh.b, acc[t]);
            acc[t] = WMMA_BF16(ahi, bl.b, acc[t]);
            acc[t] = WMMA_BF16(alo, bh.b, acc[t]);
        }
    }
}

// Kernel 1: h = xf @ W.  rows < 1024 -> h0 scratch; rows >= 1024 -> out + bias.
__global__ void __launch_bounds__(128) gat_gemm1(
    const float* __restrict__ x, const unsigned int* __restrict__ wpack,
    const float* __restrict__ bias, float* __restrict__ out, float* __restrict__ h0)
{
    const int wave  = (blockIdx.x * blockDim.x + threadIdx.x) >> 5;
    const int row0  = wave * 16;
    const int lane  = threadIdx.x & 31;
    const int half  = lane >> 4, mlane = lane & 15;
    const float* __restrict__ arow = x + (row0 + mlane) * F_DIM;

    v8f acc[NTILES];
    wave_gemm_packed(arow, F_DIM, NKT_W, wpack, lane, acc);

    const bool first = (row0 < NNODES);          // strips never straddle the boundary
#pragma unroll
    for (int t = 0; t < NTILES; ++t) {
        int col = t * 16 + mlane;
        if (col >= F_DIM) continue;
        if (first) {
#pragma unroll
            for (int r = 0; r < 8; ++r)
                h0[(row0 + half * 8 + r) * F_DIM + col] = acc[t][r];
        } else {
            float b = bias[col];
#pragma unroll
            for (int r = 0; r < 8; ++r)
                out[(row0 + half * 8 + r) * F_DIM + col] = acc[t][r] + b;
        }
    }
}

// Kernel 2: a_s = h0 @ att_src, a_d = h0 @ att_dst  (one wave32 per row).
__global__ void __launch_bounds__(32) gat_attn_coef(
    const float* __restrict__ h0,
    const float* __restrict__ att_src, const float* __restrict__ att_dst,
    float* __restrict__ a_s, float* __restrict__ a_d)
{
    int i = blockIdx.x;
    int lane = threadIdx.x;
    float ss = 0.f, sd = 0.f;
    for (int c = lane; c < F_DIM; c += 32) {
        float h = h0[i * F_DIM + c];
        ss += h * att_src[c];
        sd += h * att_dst[c];
    }
#pragma unroll
    for (int off = 16; off > 0; off >>= 1) {
        ss += __shfl_xor(ss, off, 32);
        sd += __shfl_xor(sd, off, 32);
    }
    if (lane == 0) { a_s[i] = ss; a_d[i] = sd; }
}

// Kernel 3: alpha[i,:] = softmax_j( leaky_relu(a_d[i] + a_s[j]) )
__global__ void __launch_bounds__(256) gat_softmax(
    const float* __restrict__ a_s, const float* __restrict__ a_d,
    float* __restrict__ alpha)
{
    __shared__ float red[256];
    int i = blockIdx.x;
    int tid = threadIdx.x;
    float ad = a_d[i];
    float sv[4];
    float m = -3.402823466e38f;
#pragma unroll
    for (int u = 0; u < 4; ++u) {
        int j = tid + u * 256;
        float s = ad + a_s[j];
        s = (s > 0.f) ? s : 0.2f * s;            // leaky_relu, slope 0.2
        sv[u] = s;
        m = fmaxf(m, s);
    }
    red[tid] = m; __syncthreads();
    for (int off = 128; off > 0; off >>= 1) {
        if (tid < off) red[tid] = fmaxf(red[tid], red[tid + off]);
        __syncthreads();
    }
    m = red[0]; __syncthreads();

    float sum = 0.f, ev[4];
#pragma unroll
    for (int u = 0; u < 4; ++u) { ev[u] = __expf(sv[u] - m); sum += ev[u]; }
    red[tid] = sum; __syncthreads();
    for (int off = 128; off > 0; off >>= 1) {
        if (tid < off) red[tid] += red[tid + off];
        __syncthreads();
    }
    float inv = 1.0f / red[0];
#pragma unroll
    for (int u = 0; u < 4; ++u) alpha[i * NNODES + tid + u * 256] = ev[u] * inv;
}

// Kernel 4: out0 = alpha @ h0 + bias  (rows 0..1023), h0 pre-packed.
__global__ void __launch_bounds__(128) gat_gemm2(
    const float* __restrict__ alpha, const unsigned int* __restrict__ h0pack,
    const float* __restrict__ bias, float* __restrict__ out)
{
    const int wave  = (blockIdx.x * blockDim.x + threadIdx.x) >> 5;
    const int row0  = wave * 16;
    const int lane  = threadIdx.x & 31;
    const int half  = lane >> 4, mlane = lane & 15;
    const float* __restrict__ arow = alpha + (row0 + mlane) * NNODES;

    v8f acc[NTILES];
    wave_gemm_packed(arow, NNODES, NKT_H, h0pack, lane, acc);

#pragma unroll
    for (int t = 0; t < NTILES; ++t) {
        int col = t * 16 + mlane;
        if (col >= F_DIM) continue;
        float b = bias[col];
#pragma unroll
        for (int r = 0; r < 8; ++r)
            out[(row0 + half * 8 + r) * F_DIM + col] = acc[t][r] + b;
    }
}

extern "C" void kernel_launch(void* const* d_in, const int* in_sizes, int n_in,
                              void* d_out, int out_size, void* d_ws, size_t ws_size,
                              hipStream_t stream) {
    const float* x       = (const float*)d_in[0];  // (64,1,1024,200)
    const float* W       = (const float*)d_in[1];  // (200,200)
    const float* att_src = (const float*)d_in[2];  // (200,)
    const float* att_dst = (const float*)d_in[3];  // (200,)
    const float* bias    = (const float*)d_in[4];  // (200,)
    float* out = (float*)d_out;

    // ws layout (floats/dwords):
    //   h0 (1024*200) | a_s (1024) | a_d (1024) | alpha (1024*1024) | wpack | h0pack
    float* h0    = (float*)d_ws;
    float* a_s   = h0  + NNODES * F_DIM;
    float* a_d   = a_s + NNODES;
    float* alpha = a_d + NNODES;
    unsigned int* wpack  = (unsigned int*)(alpha + NNODES * NNODES);       // 7*13*32*16 dwords
    unsigned int* h0pack = wpack + NKT_W * NTILES * 32 * 16;               // 32*13*32*16 dwords

    // Pack W into bf16 hi/lo WMMA fragments (91 fragments).
    pack_b_frags<<<NKT_W * NTILES, 32, 0, stream>>>(W, F_DIM, F_DIM, F_DIM, wpack);
    // GEMM1: 65536/16 = 4096 strips, 4 waves/block.
    gat_gemm1<<<1024, 128, 0, stream>>>(x, wpack, bias, out, h0);
    // Pack h0 for GEMM2 (416 fragments).
    pack_b_frags<<<NKT_H * NTILES, 32, 0, stream>>>(h0, F_DIM, NNODES, F_DIM, h0pack);
    gat_attn_coef<<<NNODES, 32, 0, stream>>>(h0, att_src, att_dst, a_s, a_d);
    gat_softmax<<<NNODES, 256, 0, stream>>>(a_s, a_d, alpha);
    // GEMM2: 1024/16 = 64 strips.
    gat_gemm2<<<16, 128, 0, stream>>>(alpha, h0pack, bias, out);
}
